// Net_40132174414212
// MI455X (gfx1250) — matrix operand order repaired and checked by reading, not compiled, not run on hardware
//
#include <hip/hip_runtime.h>
#include <hip/hip_bf16.h>

// ---------------------------------------------------------------------------
// SNN forward:
//   [pack w1,w2 -> f16 WMMA-B layout] -> GEMM1 (WMMA f16->f32, async-staged
//   double-buffered LDS B slabs) -> LIF scan1 -> GEMM2 (WMMA) -> LIF scan2
// ---------------------------------------------------------------------------

typedef __attribute__((ext_vector_type(16))) _Float16 v16h;
typedef __attribute__((ext_vector_type(8)))  _Float16 v8h;
typedef __attribute__((ext_vector_type(8)))  float    v8f;
typedef __attribute__((ext_vector_type(4)))  int      i4;

#define NUM_STEPS 200
#define BATCH     1024
#define MROWS     (NUM_STEPS * BATCH)   // 204800
#define NT1       19                    // ceil(300/16) N-tiles, layer 1
#define KT1       25                    // ceil(784/32) K-tiles, layer 1
#define LD1       (NT1 * 16)            // 304 (CUR1 leading dim)
#define KT2       10                    // 320/32 K-tiles, layer 2
#define LDK2      (KT2 * 32)            // 320 (SPK1 leading dim, zero padded)
#define BETA1     0.9f
#define BETA2     0.8f
#define SLAB_H    (NT1 * 512)           // f16 elems per K-slab (9728 = 19456 B)
#define SLAB_CHUNKS ((NT1 * 512 * 2) / 16)   // 1216 x 16B chunks

// ---- workspace layout (bytes). Requires ~376 MB of d_ws. ----
#define OFF_BP1   ((size_t)0)                       // 19*25*512 f16 = 486,400 B
#define OFF_BP2   ((size_t)(512 * 1024))            // 10*512 f16   =  10,240 B
#define OFF_CUR1  ((size_t)(1024 * 1024))           // 204800*304 f32 = 249,036,800 B
#define OFF_SPK1  (OFF_CUR1 + (size_t)MROWS * LD1 * 4)        // 204800*320 f16
#define OFF_CUR2  (OFF_SPK1 + (size_t)MROWS * LDK2 * 2)       // 204800*16 f32

// ---- CDNA5 async global->LDS path (guarded; falls back to sync copy) ----
#if __has_builtin(__builtin_amdgcn_global_load_async_to_lds_b128)
#define ASYNC_STAGE 1
// Builtin signature (from hipcc diagnostic): param0 = AS1 'int __vector(4) *'
typedef __attribute__((__vector_size__(16))) int      v4i_raw;
typedef __attribute__((address_space(1)))    v4i_raw  as1_v4i;
typedef __attribute__((address_space(3)))    v4i_raw  as3_v4i;
#if __has_builtin(__builtin_amdgcn_s_wait_asynccnt)
#define WAIT_ASYNC() __builtin_amdgcn_s_wait_asynccnt(0)
#else
#define WAIT_ASYNC() asm volatile("s_wait_asynccnt 0" ::: "memory")
#endif
#else
#define ASYNC_STAGE 0
#endif

// ---------------------------------------------------------------------------
// Pack w1 (f32 [300,784]) into f16 WMMA-B tiles, layout [t][j][lane][16].
// B layout per 16x16x32 tile: lane l -> column N = l%16, K = 8*(l/16) + i%8 + 16*(i/8).
// Zero pad N>=300, K>=784.
// ---------------------------------------------------------------------------
__global__ __launch_bounds__(256) void snn_pack_w1(const float* __restrict__ w1,
                                                   _Float16* __restrict__ bp1) {
    int gid  = blockIdx.x * 256 + threadIdx.x;     // grid sized exactly 243200
    int i    = gid & 15;
    int l    = (gid >> 4) & 31;
    int jt   = gid >> 9;                            // t*19 + j
    int j    = jt % NT1;
    int t    = jt / NT1;
    int n    = 16 * j + (l & 15);
    int k    = 32 * t + 8 * (l >> 4) + (i & 7) + 16 * (i >> 3);
    float v  = (n < 300 && k < 784) ? w1[n * 784 + k] : 0.0f;
    bp1[gid] = (_Float16)v;
}

// Pack w2 (f32 [10,300]) -> f16 tiles, layout [t][lane][16]; pad N>=10, K>=300.
__global__ __launch_bounds__(256) void snn_pack_w2(const float* __restrict__ w2,
                                                   _Float16* __restrict__ bp2) {
    int gid  = blockIdx.x * 256 + threadIdx.x;     // grid sized exactly 5120
    int i    = gid & 15;
    int l    = (gid >> 4) & 31;
    int t    = gid >> 9;
    int n    = l & 15;
    int k    = 32 * t + 8 * (l >> 4) + (i & 7) + 16 * (i >> 3);
    float v  = (n < 10 && k < 300) ? w2[n * 300 + k] : 0.0f;
    bp2[gid] = (_Float16)v;
}

// ---------------------------------------------------------------------------
// Stage one 19456-byte B slab (K step t) into an LDS buffer, block-wide.
// ---------------------------------------------------------------------------
__device__ __forceinline__ void stage_slab(const _Float16* __restrict__ bp1,
                                           _Float16* __restrict__ buf,
                                           int t, int tid) {
#if ASYNC_STAGE
    as1_v4i* src = (as1_v4i*)(bp1 + (size_t)t * SLAB_H);
    as3_v4i* dst = (as3_v4i*)buf;
    for (int idx = tid; idx < SLAB_CHUNKS; idx += 256)
        __builtin_amdgcn_global_load_async_to_lds_b128(src + idx, dst + idx, 0, 0);
#else
    const i4* src = (const i4*)(bp1 + (size_t)t * SLAB_H);
    i4*       dst = (i4*)buf;
    for (int idx = tid; idx < SLAB_CHUNKS; idx += 256)
        dst[idx] = src[idx];
#endif
}

// ---------------------------------------------------------------------------
// GEMM1: CUR1[204800,304] = X[204800,784](f32, cvt->f16) @ W1^T (packed f16)
// One wave per 16-row M-tile, all 19 N-tiles accumulated in registers.
// B slabs double-buffered in LDS; slab t+1 staged while computing slab t.
// ---------------------------------------------------------------------------
__global__ __launch_bounds__(256) void snn_gemm1(const float* __restrict__ X,
                                                 const _Float16* __restrict__ bp1,
                                                 float* __restrict__ cur1) {
    __shared__ alignas(32) _Float16 ldsB[2][SLAB_H];   // 2 x 19456 bytes

    const int tid  = threadIdx.x;
    const int lane = tid & 31;
    const int half = lane >> 4;
    const int m    = lane & 15;
    const int wave = blockIdx.x * 8 + (tid >> 5);      // M-tile index, 0..12799

    const float* xrow = X + (size_t)(wave * 16 + m) * 784;

    v8f acc[NT1] = {};

    // prologue: stage slab 0 into buffer 0
    stage_slab(bp1, &ldsB[0][0], 0, tid);
#if ASYNC_STAGE
    WAIT_ASYNC();
#endif
    __syncthreads();

    for (int t = 0; t < KT1; ++t) {
        const int cur = t & 1;
        // ---- kick off staging of the next slab into the other buffer ----
        if (t + 1 < KT1)
            stage_slab(bp1, &ldsB[cur ^ 1][0], t + 1, tid);

        // ---- prefetch the A stream two K-steps ahead ----
        if (t + 2 < KT1)
            __builtin_prefetch(xrow + 32 * (t + 2), 0, 0);

        // ---- load + convert A fragment (16x32 f16, ISA A-layout) ----
        // lane<16: K {0..7} in a[0..7], {16..23} in a[8..15]
        // lane>=16: K {8..15} in a[0..7], {24..31} in a[8..15]
        v8f c1 = *(const v8f*)(xrow + 32 * t + 8 * half);
        v8h h1 = __builtin_convertvector(c1, v8h);
        v8h h2 = {};
        if (t < KT1 - 1) {                 // tail tile: K 784..799 are zero
            v8f c2 = *(const v8f*)(xrow + 32 * t + 16 + 8 * half);
            h2 = __builtin_convertvector(c2, v8h);
        }
        v16h a = __builtin_shufflevector(h1, h2, 0, 1, 2, 3, 4, 5, 6, 7,
                                                 8, 9, 10, 11, 12, 13, 14, 15);

        // ---- 19 WMMAs; B tiles rotated through a 2-deep register buffer ----
        const _Float16* lb = &ldsB[cur][0];
        v16h bc = *(const v16h*)(lb + (lane << 4));          // tile 0
#pragma unroll
        for (int j = 0; j < NT1; ++j) {
            v16h bn = bc;
            if (j + 1 < NT1)                                  // load ahead
                bn = *(const v16h*)(lb + (((j + 1) * 32 + lane) << 4));
            acc[j] = __builtin_amdgcn_wmma_f32_16x16x32_f16(
                false, a, false, bc, (short)0, acc[j], false, false);
            bc = bn;
        }

        // ---- close the iteration: staged data complete + all waves done ----
#if ASYNC_STAGE
        if (t + 1 < KT1)
            WAIT_ASYNC();
#endif
        __syncthreads();
    }

    // ---- store C: lane<16 -> rows 0..7, lane>=16 -> rows 8..15; col=lane%16
    const int    col     = lane & 15;
    const size_t rowbase = (size_t)wave * 16 + half * 8;
#pragma unroll
    for (int j = 0; j < NT1; ++j) {
        float* outp = cur1 + rowbase * LD1 + j * 16 + col;
#pragma unroll
        for (int r = 0; r < 8; ++r)
            outp[(size_t)r * LD1] = acc[j][r];
    }
}

// ---------------------------------------------------------------------------
// LIF scan, layer 1: sequential over 200 steps, parallel over 1024x320 lanes.
// Writes SPK1 as f16 (spikes are exactly 0/1); columns 300..319 zero-padded.
// ---------------------------------------------------------------------------
__global__ __launch_bounds__(256) void snn_scan1(const float* __restrict__ cur1,
                                                 const float* __restrict__ b1,
                                                 _Float16* __restrict__ spk1) {
    int gid = blockIdx.x * 256 + threadIdx.x;      // grid exactly 1024*320
    int b   = gid / LDK2;
    int n   = gid % LDK2;

    if (n < 300) {
        const float bn  = b1[n];
        float       mem = 0.0f;
        for (int t = 0; t < NUM_STEPS; ++t) {
            size_t row   = (size_t)t * BATCH + b;
            float  cur   = cur1[row * LD1 + n] + bn;
            float  reset = (mem > 1.0f) ? 1.0f : 0.0f;
            mem = BETA1 * mem + cur - reset;
            spk1[row * LDK2 + n] = (_Float16)((mem > 1.0f) ? 1.0f : 0.0f);
        }
    } else {
        for (int t = 0; t < NUM_STEPS; ++t)
            spk1[((size_t)t * BATCH + b) * LDK2 + n] = (_Float16)0.0f;
    }
}

// ---------------------------------------------------------------------------
// GEMM2: CUR2[204800,16] = SPK1[204800,320](f16) @ W2^T (packed f16)
// One wave per M-tile, single N-tile, 10 K-tiles; w2 staged in LDS once.
// ---------------------------------------------------------------------------
__global__ __launch_bounds__(256) void snn_gemm2(const _Float16* __restrict__ spk1,
                                                 const _Float16* __restrict__ bp2,
                                                 float* __restrict__ cur2) {
    __shared__ alignas(32) _Float16 ldsB[KT2 * 512];   // 10240 bytes

    const int tid = threadIdx.x;
#if ASYNC_STAGE
    {
        as1_v4i* src = (as1_v4i*)bp2;
        as3_v4i* dst = (as3_v4i*)&ldsB[0];
        for (int idx = tid; idx < (KT2 * 512 * 2) / 16; idx += 256)
            __builtin_amdgcn_global_load_async_to_lds_b128(src + idx, dst + idx, 0, 0);
        WAIT_ASYNC();
    }
#else
    {
        const i4* src = (const i4*)bp2;
        i4*       dst = (i4*)ldsB;
        for (int idx = tid; idx < (KT2 * 512 * 2) / 16; idx += 256)
            dst[idx] = src[idx];
    }
#endif
    __syncthreads();

    const int lane = tid & 31;
    const int half = lane >> 4;
    const int m    = lane & 15;
    const int wave = blockIdx.x * 8 + (tid >> 5);      // 0..12799

    const _Float16* arow = spk1 + (size_t)(wave * 16 + m) * LDK2;

    v8f acc = {};
#pragma unroll
    for (int t = 0; t < KT2; ++t) {
        v8h h1 = *(const v8h*)(arow + 32 * t + 8 * half);
        v8h h2 = *(const v8h*)(arow + 32 * t + 16 + 8 * half);
        v16h a = __builtin_shufflevector(h1, h2, 0, 1, 2, 3, 4, 5, 6, 7,
                                                 8, 9, 10, 11, 12, 13, 14, 15);
        v16h b = *(const v16h*)(ldsB + ((t * 32 + lane) << 4));
        acc = __builtin_amdgcn_wmma_f32_16x16x32_f16(
            false, a, false, b, (short)0, acc, false, false);
    }

    const int    col     = lane & 15;
    const size_t rowbase = (size_t)wave * 16 + half * 8;
#pragma unroll
    for (int r = 0; r < 8; ++r)
        cur2[(rowbase + r) * 16 + col] = acc[r];
}

// ---------------------------------------------------------------------------
// LIF scan, layer 2 -> final outputs: spk2_rec[200,1024,10] ++ mem2_rec.
// ---------------------------------------------------------------------------
__global__ __launch_bounds__(256) void snn_scan2(const float* __restrict__ cur2,
                                                 const float* __restrict__ b2,
                                                 float* __restrict__ out) {
    int gid = blockIdx.x * 256 + threadIdx.x;      // grid exactly 1024*10
    int b   = gid / 10;
    int n   = gid % 10;

    const float bn  = b2[n];
    float       mem = 0.0f;
    float* out_spk = out;
    float* out_mem = out + (size_t)NUM_STEPS * BATCH * 10;   // 2,048,000
    for (int t = 0; t < NUM_STEPS; ++t) {
        float cur   = cur2[((size_t)t * BATCH + b) * 16 + n] + bn;
        float reset = (mem > 1.0f) ? 1.0f : 0.0f;
        mem = BETA2 * mem + cur - reset;
        float spk = (mem > 1.0f) ? 1.0f : 0.0f;
        out_spk[(size_t)t * (BATCH * 10) + gid] = spk;
        out_mem[(size_t)t * (BATCH * 10) + gid] = mem;
    }
}

// ---------------------------------------------------------------------------
extern "C" void kernel_launch(void* const* d_in, const int* in_sizes, int n_in,
                              void* d_out, int out_size, void* d_ws, size_t ws_size,
                              hipStream_t stream) {
    const float* X  = (const float*)d_in[0];   // [200,1024,784]
    const float* w1 = (const float*)d_in[1];   // [300,784]
    const float* b1 = (const float*)d_in[2];   // [300]
    const float* w2 = (const float*)d_in[3];   // [10,300]
    const float* b2 = (const float*)d_in[4];   // [10]
    float* out = (float*)d_out;

    char* ws = (char*)d_ws;
    _Float16* bp1  = (_Float16*)(ws + OFF_BP1);
    _Float16* bp2  = (_Float16*)(ws + OFF_BP2);
    float*    cur1 = (float*)   (ws + OFF_CUR1);
    _Float16* spk1 = (_Float16*)(ws + OFF_SPK1);
    float*    cur2 = (float*)   (ws + OFF_CUR2);

    // 1. pack weights to f16 WMMA-B layout
    snn_pack_w1<<<950, 256, 0, stream>>>(w1, bp1);   // 243200 threads
    snn_pack_w2<<<20,  256, 0, stream>>>(w2, bp2);   // 5120 threads

    // 2. batched layer-1 GEMM over all timesteps (12800 M-tiles, 8 waves/block)
    snn_gemm1<<<1600, 256, 0, stream>>>(X, bp1, cur1);

    // 3. LIF recurrence layer 1 -> spikes (f16, K-padded to 320)
    snn_scan1<<<(BATCH * LDK2) / 256, 256, 0, stream>>>(cur1, b1, spk1);

    // 4. batched layer-2 GEMM
    snn_gemm2<<<1600, 256, 0, stream>>>(spk1, bp2, cur2);

    // 5. LIF recurrence layer 2 -> outputs
    snn_scan2<<<(BATCH * 10) / 256, 256, 0, stream>>>(cur2, b2, out);
}
